// QRNN_46325517254845
// MI455X (gfx1250) — matrix operand order repaired and checked by reading, not compiled, not run on hardware
//
#include <hip/hip_runtime.h>
#include <math.h>

// Problem constants (match reference)
#define B_    8
#define T_    4096
#define CIN   512          // input channels per layer (IN == H == 512)
#define HN    512          // hidden per gate
#define G3    1536         // 3*H conv outputs
#define KR    1024         // GEMM reduction = CIN * K(=2)
#define KRP   1040         // padded K stride (2080 B rows: 16B-aligned, bank-rotated)
#define CHUNK 128          // scan chunk length
#define NCH   (T_ / CHUNK) // 32 chunks

typedef __attribute__((ext_vector_type(16))) __bf16 v16bf;
typedef __attribute__((ext_vector_type(8)))  float  v8f;
typedef __attribute__((ext_vector_type(4)))  unsigned int v4u;
typedef __attribute__((ext_vector_type(8)))  int v8i_;
typedef __attribute__((ext_vector_type(4)))  int v4i_;

__device__ __forceinline__ unsigned short f32_bf16(float f) {
  unsigned int u = __builtin_bit_cast(unsigned int, f);
  u += 0x7FFFu + ((u >> 16) & 1u);            // round-to-nearest-even
  return (unsigned short)(u >> 16);
}
__device__ __forceinline__ __bf16 us_bf(unsigned short u) {
  return __builtin_bit_cast(__bf16, u);
}
__device__ __forceinline__ float sigmoidf_(float x) {
  return 1.0f / (1.0f + __expf(-x));
}

// ---------------------------------------------------------------------------
// Weight repack: w [G3][CIN][K=2] f32 -> wbT [G3][KRP] bf16 (k-contiguous,
// r = k*CIN + c, rows padded to 1040 so LDS rows bank-rotate and stay
// 16B-aligned). GEMM A-row for token t is [x[t-1,:], x[t,:]].
// ---------------------------------------------------------------------------
__global__ __launch_bounds__(256) void prep_w(const float* __restrict__ w,
                                              unsigned short* __restrict__ wbT) {
  int idx = blockIdx.x * blockDim.x + threadIdx.x;
  if (idx >= G3 * CIN * 2) return;
  int k = idx & 1;
  int c = (idx >> 1) & (CIN - 1);
  int o = idx >> 10;                 // / (CIN*2)
  wbT[(size_t)o * KRP + (k * CIN + c)] = f32_bf16(w[idx]);
}

__global__ __launch_bounds__(256) void prep_x(const float* __restrict__ x,
                                              unsigned short* __restrict__ xb,
                                              int n) {
  int idx = blockIdx.x * blockDim.x + threadIdx.x;
  if (idx < n) xb[idx] = f32_bf16(x[idx]);
}

// ---------------------------------------------------------------------------
// Fused conv-as-GEMM + bias + gate activations.
//   * 256 threads = 8 waves; macro tile M=128 tokens x N=64 channels.
//   * Whole 64-row x 1024-k bf16 weight panel DMA'd once per block into LDS
//     by the Tensor Data Mover (plain 2-D tile, padded global rows).
//   * A tile double-buffered in LDS, staged through registers with b128
//     loads; causal k=0 tap handled by a branchless row clamp.
//   * Wave (wm 0..3, wn 0..1) owns 32x32 = 2x2 v_wmma_f32_16x16x32_bf16;
//     all fragments are contiguous 32-B groups -> ds_load_b128.
// ---------------------------------------------------------------------------
__global__ __launch_bounds__(256)
void qrnn_gemm_act(const unsigned short* __restrict__ Xb,   // [B][T][CIN] bf16
                   const unsigned short* __restrict__ WbT,  // [G3][KRP]  bf16
                   const float* __restrict__ bias,          // [G3]
                   float* __restrict__ Z, float* __restrict__ F,
                   float* __restrict__ O) {
  __shared__ unsigned short BpT[64 * KRP];        // 133120 B weight panel
  __shared__ unsigned short As[2][128][40];       // 20480 B, double buffered

  const int tid  = threadIdx.x;
  const int lane = tid & 31;
  const int wave = tid >> 5;
  const int wm = wave & 3;                 // 32-row sub-tile
  const int wn = wave >> 2;                // 32-col sub-tile
  const int nBase = blockIdx.x * 64;
  const int tBase = blockIdx.y * 128;
  const int b = blockIdx.z;

  // ---- TDM: DMA the whole [64 x 1040] bf16 weight panel into LDS ----------
#if defined(__has_builtin) && __has_builtin(__builtin_amdgcn_tensor_load_to_lds)
  if (wave == 0) {
    size_t ga = (size_t)(const void*)(WbT + (size_t)nBase * KRP); // tile start
    unsigned lds_off = (unsigned)(size_t)(void*)&BpT[0];
    // D# group 0: count=1 | lds_addr | global_addr[56:0] | type=2
    v4u g0 = {1u, lds_off, (unsigned)ga,
              (unsigned)((ga >> 32) & 0x1FFFFFFu) | (2u << 30)};
    // D# group 1: data_size=1(2B); tensor_dim0=1040, tensor_dim1=64,
    //             tile_dim0=1040, tile_dim1=64, tensor_dim0_stride=1040
    v8i_ g1 = {(int)0x00010000,
               (int)((unsigned)KRP << 16),
               (int)(64u << 16),
               (int)((unsigned)KRP << 16),
               (int)64,
               (int)KRP,
               0, 0};
    v4i_ z4 = {0, 0, 0, 0};
    v8i_ z8 = {0, 0, 0, 0, 0, 0, 0, 0};
    __builtin_amdgcn_tensor_load_to_lds(g0, g1, z4, z4, z8, 0);
    __builtin_amdgcn_s_wait_tensorcnt(0);
  }
#else
  // Fallback: cooperative copy into the same layout.
  for (int e = tid; e < 64 * KR; e += 256) {
    int n = e >> 10, k = e & (KR - 1);
    BpT[n * KRP + k] = WbT[(size_t)(nBase + n) * KRP + k];
  }
#endif
  __syncthreads();

  // ---- A-tile register staging (2 x b128 per thread per chunk) ------------
  const int arow0 = tid >> 2;              // rows 0..63
  const int arow1 = 64 + (tid >> 2);       // rows 64..127
  const int acol  = (tid & 3) * 8;         // 8 ushorts = 16 B
  uint4 ra0, ra1;

  auto loadA = [&](int c, uint4& r0, uint4& r1) {
    int k0 = c * 32;
    int dt = (k0 < CIN) ? -1 : 0;          // k=0 tap reads x[t-1]
    int xcol = (k0 & (CIN - 1)) + acol;
    int t0 = tBase + arow0 + dt;
    int t1 = tBase + arow1 + dt;
    int t0c = t0 < 0 ? 0 : t0;             // branchless clamp (only t==-1 hits)
    uint4 v0 = *(const uint4*)&Xb[((size_t)b * T_ + t0c) * CIN + xcol];
    uint4 v1 = *(const uint4*)&Xb[((size_t)b * T_ + t1) * CIN + xcol];
    if (t0 < 0) v0 = make_uint4(0, 0, 0, 0);
    r0 = v0;
    r1 = v1;
  };

  v8f acc[2][2] = {};
  const int m16  = lane & 15;
  const int kb8  = (lane < 16) ? 0 : 8;    // unused; A and B both use kb16
  const int kb16 = (lane < 16) ? 0 : 16;   // B: lane-half K base

  loadA(0, ra0, ra1);
  int buf = 0;
  for (int c = 0; c < 32; ++c) {
    // publish staged A chunk
    *(uint4*)&As[buf][arow0][acol] = ra0;
    *(uint4*)&As[buf][arow1][acol] = ra1;
    __syncthreads();
    if (c + 1 < 32) loadA(c + 1, ra0, ra1);   // overlap next global loads

    // --- A/B fragments in contiguous k order (k = 16*half + s), same for both ---
    v16bf afrag[2];
#pragma unroll
    for (int mt = 0; mt < 2; ++mt) {
      int row = 32 * wm + 16 * mt + m16;
#pragma unroll
      for (int j = 0; j < 8; ++j) {
        int k = kb16 + 2 * j;
        unsigned int u = *(const unsigned int*)&As[buf][row][k];
        afrag[mt][2 * j]     = us_bf((unsigned short)(u & 0xffffu));
        afrag[mt][2 * j + 1] = us_bf((unsigned short)(u >> 16));
      }
    }
    int kg0 = c * 32 + kb16;
#pragma unroll
    for (int nt = 0; nt < 2; ++nt) {
      int col = 32 * wn + 16 * nt + m16;
      const unsigned short* rowp = &BpT[col * KRP + kg0];
      v16bf bfrag;
#pragma unroll
      for (int j = 0; j < 8; ++j) {
        unsigned int u = *(const unsigned int*)(rowp + 2 * j);
        bfrag[2 * j]     = us_bf((unsigned short)(u & 0xffffu));
        bfrag[2 * j + 1] = us_bf((unsigned short)(u >> 16));
      }
#pragma unroll
      for (int mt = 0; mt < 2; ++mt) {
        acc[mt][nt] = __builtin_amdgcn_wmma_f32_16x16x32_bf16(
            false, afrag[mt], false, bfrag, (short)0, acc[mt][nt],
            false, false);
      }
    }
    buf ^= 1;
  }

  // --- epilogue: bias + activation, scatter to Z/F/O [B][T][H] ---
  const int rofs = (lane < 16) ? 0 : 8;
#pragma unroll
  for (int nt = 0; nt < 2; ++nt) {
    int ch   = nBase + 32 * wn + 16 * nt + m16;
    int gate = ch >> 9;                    // 0:z 1:f 2:o (64 | 512)
    int h    = ch & 511;
    float bs = bias[ch];
    float* dst = (gate == 0) ? Z : (gate == 1) ? F : O;
#pragma unroll
    for (int mt = 0; mt < 2; ++mt) {
#pragma unroll
      for (int v = 0; v < 8; ++v) {
        int t = tBase + 32 * wm + 16 * mt + rofs + v;
        float xv = acc[mt][nt][v] + bs;
        float y  = (gate == 0) ? tanhf(xv) : sigmoidf_(xv);
        dst[((size_t)b * T_ + t) * HN + h] = y;
      }
    }
  }
}

// ---------------------------------------------------------------------------
// Scan pass 1: per (b, h, chunk) compute carry A = prod(f), Bc = local h.
// ---------------------------------------------------------------------------
__global__ __launch_bounds__(256)
void scan_pass1(const float* __restrict__ Z, const float* __restrict__ F,
                float* __restrict__ cA, float* __restrict__ cB) {
  int idx = blockIdx.x * blockDim.x + threadIdx.x;
  if (idx >= B_ * NCH * HN) return;
  int h = idx & (HN - 1);
  int c = (idx >> 9) & (NCH - 1);
  int b = idx >> 14;
  size_t base = ((size_t)b * T_ + c * CHUNK) * HN + h;
  float A = 1.0f, hl = 0.0f;
#pragma unroll 4
  for (int t = 0; t < CHUNK; ++t) {
    if (t + 16 < CHUNK) {
      __builtin_prefetch(&F[base + (size_t)(t + 16) * HN], 0, 0);
      __builtin_prefetch(&Z[base + (size_t)(t + 16) * HN], 0, 0);
    }
    float f = F[base + (size_t)t * HN];
    float z = Z[base + (size_t)t * HN];
    A *= f;
    hl = f * hl + (1.0f - f) * z;
  }
  cA[idx] = A;
  cB[idx] = hl;
}

// ---------------------------------------------------------------------------
// Scan pass 2: tiny sequential combine across 32 chunks; emits per-chunk
// incoming state and the layer's final hidden state hT.
// ---------------------------------------------------------------------------
__global__ __launch_bounds__(256)
void scan_pass2(const float* __restrict__ cA, const float* __restrict__ cB,
                float* __restrict__ hin, float* __restrict__ hT) {
  int idx = blockIdx.x * blockDim.x + threadIdx.x;
  if (idx >= B_ * HN) return;
  int h = idx & (HN - 1);
  int b = idx >> 9;
  float hs = 0.0f;
#pragma unroll
  for (int c = 0; c < NCH; ++c) {
    size_t q = ((size_t)b * NCH + c) * HN + h;
    hin[q] = hs;
    hs = cA[q] * hs + cB[q];
  }
  hT[(size_t)b * HN + h] = hs;
}

// ---------------------------------------------------------------------------
// Scan pass 3: replay chunks with correct incoming state, emit o*h.
// ---------------------------------------------------------------------------
template <bool OUT_BF16>
__global__ __launch_bounds__(256)
void scan_pass3(const float* __restrict__ Z, const float* __restrict__ F,
                const float* __restrict__ O, const float* __restrict__ hin,
                void* __restrict__ out) {
  int idx = blockIdx.x * blockDim.x + threadIdx.x;
  if (idx >= B_ * NCH * HN) return;
  int h = idx & (HN - 1);
  int c = (idx >> 9) & (NCH - 1);
  int b = idx >> 14;
  size_t base = ((size_t)b * T_ + c * CHUNK) * HN + h;
  float hc = hin[idx];
#pragma unroll 4
  for (int t = 0; t < CHUNK; ++t) {
    if (t + 16 < CHUNK) {
      __builtin_prefetch(&F[base + (size_t)(t + 16) * HN], 0, 0);
      __builtin_prefetch(&Z[base + (size_t)(t + 16) * HN], 0, 0);
      __builtin_prefetch(&O[base + (size_t)(t + 16) * HN], 0, 0);
    }
    float f = F[base + (size_t)t * HN];
    float z = Z[base + (size_t)t * HN];
    float o = O[base + (size_t)t * HN];
    hc = f * hc + (1.0f - f) * z;
    float y = o * hc;
    if (OUT_BF16)
      ((unsigned short*)out)[base + (size_t)t * HN] = f32_bf16(y);
    else
      ((float*)out)[base + (size_t)t * HN] = y;
  }
}

// ---------------------------------------------------------------------------
extern "C" void kernel_launch(void* const* d_in, const int* in_sizes, int n_in,
                              void* d_out, int out_size, void* d_ws,
                              size_t ws_size, hipStream_t stream) {
  (void)in_sizes; (void)n_in; (void)out_size; (void)ws_size;
  const float* x  = (const float*)d_in[0];
  const float* w0 = (const float*)d_in[1];
  const float* b0 = (const float*)d_in[2];
  const float* w1 = (const float*)d_in[3];
  const float* b1 = (const float*)d_in[4];

  char* p = (char*)d_ws;
  auto take = [&](size_t bytes) {
    char* r = p;
    p += (bytes + 255) & ~(size_t)255;
    return r;
  };
  unsigned short* wb0 = (unsigned short*)take((size_t)G3 * KRP * 2);
  unsigned short* wb1 = (unsigned short*)take((size_t)G3 * KRP * 2);
  unsigned short* xb  = (unsigned short*)take((size_t)B_ * T_ * CIN * 2);
  unsigned short* o0b = (unsigned short*)take((size_t)B_ * T_ * HN * 2);
  float* Z   = (float*)take((size_t)B_ * T_ * HN * 4);
  float* F   = (float*)take((size_t)B_ * T_ * HN * 4);
  float* O   = (float*)take((size_t)B_ * T_ * HN * 4);
  float* cA  = (float*)take((size_t)B_ * NCH * HN * 4);
  float* cB  = (float*)take((size_t)B_ * NCH * HN * 4);
  float* hin = (float*)take((size_t)B_ * NCH * HN * 4);

  float* out = (float*)d_out;
  float* h0  = out + (size_t)B_ * T_ * HN;
  float* h1  = h0 + (size_t)B_ * HN;

  // --- input/weight repack to bf16 ---
  prep_w<<<(G3 * KR + 255) / 256, 256, 0, stream>>>(w0, wb0);
  prep_w<<<(G3 * KR + 255) / 256, 256, 0, stream>>>(w1, wb1);
  prep_x<<<(B_ * T_ * CIN + 255) / 256, 256, 0, stream>>>(x, xb, B_ * T_ * CIN);

  dim3 gg(G3 / 64, T_ / 128, B_);
  int scanN = B_ * NCH * HN;
  int combN = B_ * HN;

  // --- layer 0 ---
  qrnn_gemm_act<<<gg, 256, 0, stream>>>(xb, wb0, b0, Z, F, O);
  scan_pass1<<<(scanN + 255) / 256, 256, 0, stream>>>(Z, F, cA, cB);
  scan_pass2<<<(combN + 255) / 256, 256, 0, stream>>>(cA, cB, hin, h0);
  scan_pass3<true><<<(scanN + 255) / 256, 256, 0, stream>>>(Z, F, O, hin, o0b);

  // --- layer 1 ---
  qrnn_gemm_act<<<gg, 256, 0, stream>>>(o0b, wb1, b1, Z, F, O);
  scan_pass1<<<(scanN + 255) / 256, 256, 0, stream>>>(Z, F, cA, cB);
  scan_pass2<<<(combN + 255) / 256, 256, 0, stream>>>(cA, cB, hin, h1);
  scan_pass3<false><<<(scanN + 255) / 256, 256, 0, stream>>>(Z, F, O, hin, out);
}
